// GraphSAGE_31799937859849
// MI455X (gfx1250) — matrix operand order repaired and checked by reading, not compiled, not run on hardware
//
#include <hip/hip_runtime.h>
#include <hip/hip_bf16.h>

#define N_NODES 50000
#define N_PAD   50048            // multiple of 64: guard-free M tiling
#define DEG 16
#define WIDTH 256
#define BATCH 8192
#define MB32 (N_PAD / 32)        // 1564 exact

typedef __attribute__((ext_vector_type(16))) __bf16        v16bf;
typedef __attribute__((ext_vector_type(8)))  float         v8f;
typedef __attribute__((ext_vector_type(4)))  unsigned int  u32x4;

static __device__ __forceinline__ unsigned short f2bf(float x) {
    unsigned u = __float_as_uint(x);
    u += 0x7fffu + ((u >> 16) & 1u);          // round-to-nearest-even
    return (unsigned short)(u >> 16);
}
static __device__ __forceinline__ float bf2f(unsigned short h) {
    return __uint_as_float(((unsigned)h) << 16);
}
static __device__ __forceinline__ v16bf load16bf(const unsigned short* p0,
                                                 const unsigned short* p1) {
    union { v16bf v; u32x4 q[2]; } u;
    u.q[0] = *reinterpret_cast<const u32x4*>(p0);
    u.q[1] = *reinterpret_cast<const u32x4*>(p1);
    return u.v;
}

// ---------------- init / scatter ----------------
__global__ void k_zero(float* __restrict__ H32, unsigned short* __restrict__ Hb) {
    int i = blockIdx.x * 256 + threadIdx.x;   // N_PAD*WIDTH total
    H32[i] = 0.0f;
    Hb[i]  = 0;
}
__global__ void k_zero_pad(unsigned short* __restrict__ NMb) {
    int i = blockIdx.x * 256 + threadIdx.x;   // (N_PAD-N_NODES)*WIDTH total
    NMb[(size_t)N_NODES * WIDTH + i] = 0;
}
__global__ void k_winner_init(int* __restrict__ winner) {
    int i = blockIdx.x * 256 + threadIdx.x;
    if (i < N_NODES) winner[i] = -1;
}
__global__ void k_winner(const int* __restrict__ node_idx, int* __restrict__ winner) {
    int b = blockIdx.x * 256 + threadIdx.x;
    if (b < BATCH) atomicMax(&winner[node_idx[b]], b);
}
__global__ void k_scatter(const int* __restrict__ node_idx,
                          const float* __restrict__ feats,
                          const int* __restrict__ winner,
                          float* __restrict__ H32, unsigned short* __restrict__ Hb) {
    int i = blockIdx.x * 256 + threadIdx.x;   // BATCH*WIDTH total
    int b = i >> 8, c = i & 255;
    int row = node_idx[b];
    if (winner[row] != b) return;             // deterministic duplicate resolution
    float f = feats[i];
    H32[(size_t)row * WIDTH + c] = f;
    Hb [(size_t)row * WIDTH + c] = f2bf(f);
}

// ---------------- weight packing into WMMA B-tile layout ----------------
// tiles (kc-major, nt-minor), 512 bf16/tile; lane l: col N = nt*16+(l&15),
// element j: K = kc*32 + (l>>4)*16 + j  (contiguous 32B per lane).
__global__ void k_pack(const float* __restrict__ W, unsigned short* __restrict__ dst) {
    int t = blockIdx.x * 256 + threadIdx.x;   // Ktot*256 total
    int j    = t & 15;
    int lane = (t >> 4) & 31;
    int tile = t >> 9;
    int nt = tile & 15;
    int kc = tile >> 4;
    int n = nt * 16 + (lane & 15);
    int k = kc * 32 + (lane >> 4) * 16 + j;
    dst[t] = f2bf(W[(size_t)k * WIDTH + n]);
}

// ---------------- GEMM 1: AGG = relu(H @ W_agg + b_agg) ----------------
// One wave: 32 rows x 64 cols. Each B tile feeds 2 WMMAs.
__global__ void k_gemm_agg(const unsigned short* __restrict__ Hb,
                           const unsigned short* __restrict__ Wp,
                           const float* __restrict__ bias,
                           unsigned short* __restrict__ AGGb) {
    int wid  = threadIdx.x >> 5;
    int lane = threadIdx.x & 31;
    int task = blockIdx.x * 8 + wid;          // MB32*4 tasks, exact
    int mblk = task >> 2;
    int ng   = task & 3;
    int lo = lane & 15, hi = lane >> 4;
    int m0 = mblk * 32;
    int n0 = ng * 64;

    v8f acc[2][4];
#pragma unroll
    for (int nt = 0; nt < 4; ++nt) {
        float bv = bias[n0 + nt * 16 + lo];
        v8f a = {bv, bv, bv, bv, bv, bv, bv, bv};
        acc[0][nt] = a;
        acc[1][nt] = a;
    }

    const unsigned short* arow0 = Hb + (size_t)(m0 + lo) * WIDTH;
    const unsigned short* arow1 = arow0 + (size_t)16 * WIDTH;
#pragma unroll
    for (int kc = 0; kc < 8; ++kc) {
        int ko = kc * 32 + hi * 16;
        v16bf a0 = load16bf(arow0 + ko, arow0 + ko + 8);
        v16bf a1 = load16bf(arow1 + ko, arow1 + ko + 8);
        v16bf b0, b1, b2, b3;
        {
            const unsigned short* bp = Wp + ((size_t)(kc * 16 + ng * 4) * 512) + lane * 16;
            b0 = load16bf(bp, bp + 8);
            b1 = load16bf(bp + 512, bp + 512 + 8);
            b2 = load16bf(bp + 1024, bp + 1024 + 8);
            b3 = load16bf(bp + 1536, bp + 1536 + 8);
        }
        acc[0][0] = __builtin_amdgcn_wmma_f32_16x16x32_bf16(false, a0, false, b0, (short)0, acc[0][0], false, false);
        acc[1][0] = __builtin_amdgcn_wmma_f32_16x16x32_bf16(false, a1, false, b0, (short)0, acc[1][0], false, false);
        acc[0][1] = __builtin_amdgcn_wmma_f32_16x16x32_bf16(false, a0, false, b1, (short)0, acc[0][1], false, false);
        acc[1][1] = __builtin_amdgcn_wmma_f32_16x16x32_bf16(false, a1, false, b1, (short)0, acc[1][1], false, false);
        acc[0][2] = __builtin_amdgcn_wmma_f32_16x16x32_bf16(false, a0, false, b2, (short)0, acc[0][2], false, false);
        acc[1][2] = __builtin_amdgcn_wmma_f32_16x16x32_bf16(false, a1, false, b2, (short)0, acc[1][2], false, false);
        acc[0][3] = __builtin_amdgcn_wmma_f32_16x16x32_bf16(false, a0, false, b3, (short)0, acc[0][3], false, false);
        acc[1][3] = __builtin_amdgcn_wmma_f32_16x16x32_bf16(false, a1, false, b3, (short)0, acc[1][3], false, false);
    }

#pragma unroll
    for (int mi = 0; mi < 2; ++mi)
#pragma unroll
        for (int nt = 0; nt < 4; ++nt)
#pragma unroll
            for (int r = 0; r < 8; ++r) {
                int row = m0 + mi * 16 + r + hi * 8;
                int col = n0 + nt * 16 + lo;
                float v = acc[mi][nt][r];
                AGGb[(size_t)row * WIDTH + col] = f2bf(v > 0.0f ? v : 0.0f);
            }
}

// ---------------- neighbor max over DEG gathered rows ----------------
__global__ void k_gathermax(const unsigned short* __restrict__ AGGb,
                            const int* __restrict__ nbd,
                            unsigned short* __restrict__ NMb) {
    __shared__ int idx[DEG];
    int n = blockIdx.x;
    if (threadIdx.x < DEG) idx[threadIdx.x] = nbd[(size_t)n * DEG + threadIdx.x];
    __syncthreads();
    int c = threadIdx.x;
    float m = -3.4e38f;
#pragma unroll
    for (int d = 0; d < DEG; ++d)
        m = fmaxf(m, bf2f(AGGb[(size_t)idx[d] * WIDTH + c]));
    NMb[(size_t)n * WIDTH + c] = f2bf(m);
}

// ---------------- GEMM 2 + normalize: H = normalize(relu([H|NM] @ W + b)) ----------------
// Block: 4 waves, 32 rows x full 256 cols; each wave 2 Mtiles x 4 Ntiles.
__global__ void k_gemm_lin(const unsigned short* __restrict__ Hb,
                           const unsigned short* __restrict__ NMb,
                           const unsigned short* __restrict__ Wp,
                           const float* __restrict__ bias,
                           float* __restrict__ Hout32,
                           unsigned short* __restrict__ Houtb) {
    __shared__ float tile[32][WIDTH];
    __shared__ float inv[32];
    int lane = threadIdx.x & 31;
    int ng   = threadIdx.x >> 5;              // 0..3
    int lo = lane & 15, hi = lane >> 4;
    int m0 = blockIdx.x * 32;
    int n0 = ng * 64;

    v8f acc[2][4];
#pragma unroll
    for (int nt = 0; nt < 4; ++nt) {
        float bv = bias[n0 + nt * 16 + lo];
        v8f a = {bv, bv, bv, bv, bv, bv, bv, bv};
        acc[0][nt] = a;
        acc[1][nt] = a;
    }

#pragma unroll
    for (int kc = 0; kc < 16; ++kc) {         // K=512: first 256 from H, then NM
        const unsigned short* src = (kc < 8) ? Hb : NMb;
        int ko = (kc & 7) * 32 + hi * 16;
        const unsigned short* arow0 = src + (size_t)(m0 + lo) * WIDTH + ko;
        const unsigned short* arow1 = arow0 + (size_t)16 * WIDTH;
        v16bf a0 = load16bf(arow0, arow0 + 8);
        v16bf a1 = load16bf(arow1, arow1 + 8);
        v16bf b0, b1, b2, b3;
        {
            const unsigned short* bp = Wp + ((size_t)(kc * 16 + ng * 4) * 512) + lane * 16;
            b0 = load16bf(bp, bp + 8);
            b1 = load16bf(bp + 512, bp + 512 + 8);
            b2 = load16bf(bp + 1024, bp + 1024 + 8);
            b3 = load16bf(bp + 1536, bp + 1536 + 8);
        }
        acc[0][0] = __builtin_amdgcn_wmma_f32_16x16x32_bf16(false, a0, false, b0, (short)0, acc[0][0], false, false);
        acc[1][0] = __builtin_amdgcn_wmma_f32_16x16x32_bf16(false, a1, false, b0, (short)0, acc[1][0], false, false);
        acc[0][1] = __builtin_amdgcn_wmma_f32_16x16x32_bf16(false, a0, false, b1, (short)0, acc[0][1], false, false);
        acc[1][1] = __builtin_amdgcn_wmma_f32_16x16x32_bf16(false, a1, false, b1, (short)0, acc[1][1], false, false);
        acc[0][2] = __builtin_amdgcn_wmma_f32_16x16x32_bf16(false, a0, false, b2, (short)0, acc[0][2], false, false);
        acc[1][2] = __builtin_amdgcn_wmma_f32_16x16x32_bf16(false, a1, false, b2, (short)0, acc[1][2], false, false);
        acc[0][3] = __builtin_amdgcn_wmma_f32_16x16x32_bf16(false, a0, false, b3, (short)0, acc[0][3], false, false);
        acc[1][3] = __builtin_amdgcn_wmma_f32_16x16x32_bf16(false, a1, false, b3, (short)0, acc[1][3], false, false);
    }

#pragma unroll
    for (int mi = 0; mi < 2; ++mi)
#pragma unroll
        for (int nt = 0; nt < 4; ++nt)
#pragma unroll
            for (int r = 0; r < 8; ++r) {
                float v = acc[mi][nt][r];
                tile[mi * 16 + r + hi * 8][n0 + nt * 16 + lo] = v > 0.0f ? v : 0.0f;
            }
    __syncthreads();

    if (threadIdx.x < 32) {
        float s = 0.0f;
        for (int c = 0; c < WIDTH; ++c) {
            float v = tile[threadIdx.x][c];
            s += v * v;
        }
        inv[threadIdx.x] = 1.0f / fmaxf(sqrtf(s), 1e-12f);
    }
    __syncthreads();

    int row = threadIdx.x >> 2;               // 0..31
    int cb  = (threadIdx.x & 3) * 64;
    float iv = inv[row];
    size_t gbase = (size_t)(m0 + row) * WIDTH;
    for (int c = cb; c < cb + 64; ++c) {
        float v = tile[row][c] * iv;
        Hout32[gbase + c] = v;
        Houtb [gbase + c] = f2bf(v);          // in-place on Hb: block-local rows only
    }
}

// ---------------- final gather ----------------
__global__ void k_gather_out(const int* __restrict__ node_idx,
                             const float* __restrict__ H32,
                             float* __restrict__ out) {
    int i = blockIdx.x * 256 + threadIdx.x;   // BATCH*WIDTH
    int b = i >> 8, c = i & 255;
    out[i] = H32[(size_t)node_idx[b] * WIDTH + c];
}

extern "C" void kernel_launch(void* const* d_in, const int* in_sizes, int n_in,
                              void* d_out, int out_size, void* d_ws, size_t ws_size,
                              hipStream_t stream) {
    const int*   node_idx = (const int*)  d_in[0];
    const float* feats    = (const float*)d_in[1];
    const int*   nbd      = (const int*)  d_in[2];
    const float* W_agg    = (const float*)d_in[3];
    const float* b_agg    = (const float*)d_in[4];
    const float* W_lin    = (const float*)d_in[5];
    const float* b_lin    = (const float*)d_in[6];
    float* out = (float*)d_out;

    // workspace carve (all 256-B aligned; padded to N_PAD rows)
    char* ws = (char*)d_ws;
    float*          H32  = (float*)ws;          ws += (size_t)N_PAD * WIDTH * 4;
    unsigned short* Hb   = (unsigned short*)ws; ws += (size_t)N_PAD * WIDTH * 2;
    unsigned short* AGGb = (unsigned short*)ws; ws += (size_t)N_PAD * WIDTH * 2;
    unsigned short* NMb  = (unsigned short*)ws; ws += (size_t)N_PAD * WIDTH * 2;
    unsigned short* WAp  = (unsigned short*)ws; ws += (size_t)WIDTH * WIDTH * 2;
    unsigned short* WLp  = (unsigned short*)ws; ws += (size_t)2 * 512 * WIDTH * 2;
    int*            winner = (int*)ws;

    // H^0 init (padding rows zeroed too)
    k_zero<<<(N_PAD * WIDTH) / 256, 256, 0, stream>>>(H32, Hb);
    k_zero_pad<<<((N_PAD - N_NODES) * WIDTH) / 256, 256, 0, stream>>>(NMb);
    k_winner_init<<<(N_NODES + 255) / 256, 256, 0, stream>>>(winner);
    k_winner<<<(BATCH + 255) / 256, 256, 0, stream>>>(node_idx, winner);
    k_scatter<<<(BATCH * WIDTH) / 256, 256, 0, stream>>>(node_idx, feats, winner, H32, Hb);

    // weight packing (bf16 WMMA B-tile layout)
    k_pack<<<(WIDTH * WIDTH) / 256, 256, 0, stream>>>(W_agg, WAp);
    k_pack<<<(512 * WIDTH) / 256, 256, 0, stream>>>(W_lin, WLp);
    k_pack<<<(512 * WIDTH) / 256, 256, 0, stream>>>(W_lin + (size_t)512 * WIDTH,
                                                    WLp + (size_t)512 * WIDTH);

    for (int k = 0; k < 2; ++k) {
        k_gemm_agg<<<(MB32 * 4) / 8, 256, 0, stream>>>(Hb, WAp, b_agg, AGGb);
        k_gathermax<<<N_NODES, 256, 0, stream>>>(AGGb, nbd, NMb);
        k_gemm_lin<<<MB32, 128, 0, stream>>>(Hb, NMb,
                                             WLp + (size_t)k * 512 * WIDTH,
                                             b_lin + (size_t)k * WIDTH,
                                             H32, Hb);
    }
    k_gather_out<<<BATCH, 256, 0, stream>>>(node_idx, H32, out);
}
